// ESNCell_51823075393717
// MI455X (gfx1250) — matrix operand order repaired and checked by reading, not compiled, not run on hardware
//
#include <hip/hip_runtime.h>
#include <hip/hip_bf16.h>
#include <math.h>

// Problem constants (from reference)
#define B_  32
#define T_  2048
#define I_  128
#define N_  1024
#define NWG 32          // persistent workgroups in recurrent kernel

typedef __attribute__((ext_vector_type(16))) __bf16 v16bf;
typedef __attribute__((ext_vector_type(8)))  float  v8f;

#if __has_builtin(__builtin_amdgcn_global_load_async_to_lds_b128)
#define HAVE_ASYNC_LDS 1
// Exact parameter types per hipcc diagnostic: (AS1 int4*, AS3 int4*, imm, imm)
typedef int v4i_ __attribute__((__vector_size__(4 * sizeof(int))));
typedef __attribute__((address_space(1))) v4i_* gas_ptr;
typedef __attribute__((address_space(3))) v4i_* las_ptr;
#else
#define HAVE_ASYNC_LDS 0
#endif

#if __has_builtin(__builtin_amdgcn_s_wait_asynccnt)
#define WAIT_ASYNC0() __builtin_amdgcn_s_wait_asynccnt(0)
#else
#define WAIT_ASYNC0() asm volatile("s_wait_asynccnt 0x0" ::: "memory")
#endif

union BF16Frag {
  v16bf v;
  unsigned short h[16];
  uint4 q[2];
};

__device__ __forceinline__ unsigned short f2bf(float f) {
  union { float f; unsigned u; } x; x.f = f;
  unsigned r = x.u + 0x7FFFu + ((x.u >> 16) & 1u);   // round-to-nearest-even
  return (unsigned short)(r >> 16);
}
__device__ __forceinline__ unsigned pack2bf(float a, float b) {
  return (unsigned)f2bf(a) | ((unsigned)f2bf(b) << 16);
}

// ---------------------------------------------------------------------------
// Kernel 0: convert W_in (f32 [N,I] row-major) -> bf16 row-major in workspace.
// ---------------------------------------------------------------------------
__global__ void __launch_bounds__(256) winConvertKernel(
    const float* __restrict__ w_in, unsigned* __restrict__ w_in_bf_packed) {
  const int np = (N_ * I_) / 2;
  for (int p = blockIdx.x * blockDim.x + threadIdx.x; p < np;
       p += gridDim.x * blockDim.x) {
    w_in_bf_packed[p] = pack2bf(w_in[2 * p], w_in[2 * p + 1]);
  }
}

// ---------------------------------------------------------------------------
// Kernel 1: pre[b,t,n] = u[b,t,:] @ W_in[n,:] + bias[n], written into d_out.
// [B*T x I] x [I x N] GEMM, bf16 WMMA, f32 accumulate. One wave per 16-row
// M-tile (4096 tiles), looping over the 64 N-tiles (B frags stay L2-hot).
// Bandwidth-bound: 288 MB of mandatory traffic ~ 12 us at 23.3 TB/s.
// ---------------------------------------------------------------------------
__global__ void __launch_bounds__(256) inputProjKernel(
    const float* __restrict__ u,
    const unsigned short* __restrict__ w_in_bf,
    const float* __restrict__ w_bias,
    float* __restrict__ out) {
  const int lane  = threadIdx.x & 31;
  const int wid   = threadIdx.x >> 5;
  const int mt    = blockIdx.x * 8 + wid;   // 0..4095
  const int m0    = mt * 16;
  const int l16   = lane & 15;
  const int lhalf = lane >> 4;

  // A fragments for K=128 (4 k-steps of 32), converted f32->bf16 on the fly.
  BF16Frag a[4];
  const float* urow = u + (size_t)(m0 + l16) * I_;
#pragma unroll
  for (int s = 0; s < 4; ++s) {
    const int c1 = s * 32 + lhalf * 16;
    float4 f0 = *(const float4*)(urow + c1);
    float4 f1 = *(const float4*)(urow + c1 + 4);
    float4 f2 = *(const float4*)(urow + c1 + 8);
    float4 f3 = *(const float4*)(urow + c1 + 12);
    a[s].h[0]=f2bf(f0.x);  a[s].h[1]=f2bf(f0.y);  a[s].h[2]=f2bf(f0.z);  a[s].h[3]=f2bf(f0.w);
    a[s].h[4]=f2bf(f1.x);  a[s].h[5]=f2bf(f1.y);  a[s].h[6]=f2bf(f1.z);  a[s].h[7]=f2bf(f1.w);
    a[s].h[8]=f2bf(f2.x);  a[s].h[9]=f2bf(f2.y);  a[s].h[10]=f2bf(f2.z); a[s].h[11]=f2bf(f2.w);
    a[s].h[12]=f2bf(f3.x); a[s].h[13]=f2bf(f3.y); a[s].h[14]=f2bf(f3.z); a[s].h[15]=f2bf(f3.w);
  }

  for (int nt = 0; nt < N_ / 16; ++nt) {
    const int n0 = nt * 16;
    v8f acc = {0.f, 0.f, 0.f, 0.f, 0.f, 0.f, 0.f, 0.f};
    const unsigned short* brow = w_in_bf + (size_t)(n0 + l16) * I_;
#pragma unroll
    for (int s = 0; s < 4; ++s) {
      BF16Frag b;
      const int koff = s * 32 + lhalf * 16;
      b.q[0] = *(const uint4*)(brow + koff);
      b.q[1] = *(const uint4*)(brow + koff + 8);
      acc = __builtin_amdgcn_wmma_f32_16x16x32_bf16(
          false, a[s].v, false, b.v, (short)0, acc, false, false);
    }
    const float bias = w_bias[n0 + l16];
    float* obase = out + (size_t)(m0 + lhalf * 8) * N_ + n0 + l16;
#pragma unroll
    for (int r = 0; r < 8; ++r) obase[(size_t)r * N_] = acc[r] + bias;
  }
}

// ---------------------------------------------------------------------------
// Kernel 2: persistent recurrent kernel.
//   h_t = tanh(pre_t + h_{t-1} @ W^T), pre_t already in d_out (overwritten).
// 32 WGs x 128 threads (4 waves: 2 M-tiles x 2 N-tiles of the WG's 32 cols).
// - W slice lives in LDS as pre-swizzled bf16 WMMA B-fragments (64 KB).
// - 4 independent WMMA accumulators: dependency chain 32 -> 8 XDL latencies.
// - pre_t+1 tile async-copied global->LDS (double buffered) under the WMMAs.
// - h double-buffered bf16 in workspace, broadcast via the 192 MB L2.
// ---------------------------------------------------------------------------
__global__ void __launch_bounds__(128) recurrentKernel(
    const float* __restrict__ w,
    float* __restrict__ out,
    unsigned short* __restrict__ hbuf,     // [2][B_][N_] bf16, zero-initialized
    unsigned* __restrict__ barrier_ctr) {  // zero-initialized
  __shared__ __align__(16) unsigned short wlds[32 * N_];   // 64 KB
#if HAVE_ASYNC_LDS
  __shared__ __align__(16) float preStage[4][2][256];      // 8 KB: per-wave double buffer
#endif

  const int wg    = blockIdx.x;          // 0..31 -> W rows [wg*32, wg*32+32)
  const int lane  = threadIdx.x & 31;
  const int wid   = threadIdx.x >> 5;    // 0..3
  const int mt    = wid >> 1;            // M tile (batch rows 0-15 / 16-31)
  const int j     = wid & 1;             // N tile within WG
  const int l16   = lane & 15;
  const int lhalf = lane >> 4;
  const int m0    = mt * 16;
  const int n0g   = wg * 32 + j * 16;    // global output-column base

  // ---- stage W slice into LDS, converted to bf16 in B-fragment layout.
  // Block (j,s) = 1 KB: lane L holds W[n0+j*16+(L&15), s*32 + (L>=16?16:0) + 0..15].
  // (Swizzled layout: row-major would give a 16-way LDS bank conflict.)
  {
    const int nrow0 = wg * 32;
    unsigned* wlds32 = (unsigned*)wlds;
    for (int p = threadIdx.x; p < (32 * N_) / 2; p += blockDim.x) {
      const int n_local = (2 * p) >> 10;
      const int k       = (2 * p) & (N_ - 1);
      const float* wr = w + (size_t)(nrow0 + n_local) * N_ + k;
      const unsigned val = pack2bf(wr[0], wr[1]);
      const int jj = n_local >> 4, ll = n_local & 15;
      const int s = k >> 5, kk = k & 31, sel = kk >> 4;
      const int byteoff = ((jj << 5) + s) * 1024 + (ll + (sel << 4)) * 32 + (kk & 15) * 2;
      wlds32[byteoff >> 2] = val;
    }
  }
  __syncthreads();

  const unsigned char* ldsb = (const unsigned char*)wlds;
  // Per-wave pre-activation tile base for t: out[(m0*T + t)*N + n0g], 16 rows x 64 B.
  const float* ptile0 = out + ((size_t)m0 * T_) * N_ + n0g;

#if HAVE_ASYNC_LDS
  // async-stage helper indices: 32 lanes x 16 B x 2 instrs = 1 KB tile
  const int arr = lane >> 2;             // 0..7
  const int acb = (lane & 3) * 4;        // float col offset (16 B chunks)
  {
    // prologue: stage pre tile for t = 0 into buffer 0
    const float* g = ptile0;             // t = 0
    float* sbuf = preStage[wid][0];
#pragma unroll
    for (int half = 0; half < 2; ++half) {
      const int rr = arr + half * 8;
      __builtin_amdgcn_global_load_async_to_lds_b128(
          (gas_ptr)(g + (size_t)rr * T_ * N_ + acb),
          (las_ptr)(sbuf + rr * 16 + acb), 0, 0);
    }
  }
#endif

  for (int t = 0; t < T_; ++t) {
    const unsigned short* hcur  = hbuf + (size_t)(t & 1) * (B_ * N_);
    unsigned short*       hnext = hbuf + (size_t)((t + 1) & 1) * (B_ * N_);

    // ---- acc0 <- pre-activation tile (input projection + bias)
    v8f acc0;
#if HAVE_ASYNC_LDS
    WAIT_ASYNC0();                       // staged tile for this t is ready
    {
      const float* sbuf = preStage[wid][t & 1];
#pragma unroll
      for (int r = 0; r < 8; ++r) acc0[r] = sbuf[(r + 8 * lhalf) * 16 + l16];
    }
    if (t + 1 < T_) {                    // overlap next tile with the WMMAs
      const float* g = ptile0 + (size_t)(t + 1) * N_;
      float* sbuf = preStage[wid][(t + 1) & 1];
#pragma unroll
      for (int half = 0; half < 2; ++half) {
        const int rr = arr + half * 8;
        __builtin_amdgcn_global_load_async_to_lds_b128(
            (gas_ptr)(g + (size_t)rr * T_ * N_ + acb),
            (las_ptr)(sbuf + rr * 16 + acb), 0, 0);
      }
    }
#else
    {
      const float* pbase = ptile0 + (size_t)(lhalf * 8) * T_ * N_ + (size_t)t * N_ + l16;
#pragma unroll
      for (int r = 0; r < 8; ++r) acc0[r] = pbase[(size_t)r * T_ * N_];
      if (t + 1 < T_) __builtin_prefetch(pbase + N_, 0, 1);
    }
#endif

    // ---- h_{t-1} @ W^T : 32 bf16 WMMAs over K=1024, 4 independent chains
    v8f acc1 = {0.f,0.f,0.f,0.f,0.f,0.f,0.f,0.f};
    v8f acc2 = {0.f,0.f,0.f,0.f,0.f,0.f,0.f,0.f};
    v8f acc3 = {0.f,0.f,0.f,0.f,0.f,0.f,0.f,0.f};
    const unsigned short* hrow = hcur + (size_t)(m0 + l16) * N_;
#pragma unroll 4
    for (int s = 0; s < 32; ++s) {
      BF16Frag a, b;
      const int c1 = s * 32 + lhalf * 16;
      a.q[0] = *(const uint4*)(hrow + c1);        // K c1..c1+7
      a.q[1] = *(const uint4*)(hrow + c1 + 8);    // K c1+8..c1+15
      const unsigned char* bb = ldsb + ((j << 5) + s) * 1024 + lane * 32;
      b.q[0] = *(const uint4*)(bb);
      b.q[1] = *(const uint4*)(bb + 16);
      switch (s & 3) {
        case 0: acc0 = __builtin_amdgcn_wmma_f32_16x16x32_bf16(
                    false, a.v, false, b.v, (short)0, acc0, false, false); break;
        case 1: acc1 = __builtin_amdgcn_wmma_f32_16x16x32_bf16(
                    false, a.v, false, b.v, (short)0, acc1, false, false); break;
        case 2: acc2 = __builtin_amdgcn_wmma_f32_16x16x32_bf16(
                    false, a.v, false, b.v, (short)0, acc2, false, false); break;
        default: acc3 = __builtin_amdgcn_wmma_f32_16x16x32_bf16(
                    false, a.v, false, b.v, (short)0, acc3, false, false); break;
      }
    }

    // ---- combine, tanh, writeback (out is final: non-temporal)
    {
      float* obase =
          out + ((size_t)(m0 + lhalf * 8) * T_ + t) * N_ + n0g + l16;
      unsigned short* hb = hnext + (size_t)(m0 + lhalf * 8) * N_ + n0g + l16;
#pragma unroll
      for (int r = 0; r < 8; ++r) {
        const float v = tanhf((acc0[r] + acc1[r]) + (acc2[r] + acc3[r]));
        __builtin_nontemporal_store(v, &obase[(size_t)r * T_ * N_]);
        hb[(size_t)r * N_] = f2bf(v);
      }
    }

    // ---- inter-WG sync: cluster barrier (fast path if cluster-dispatched;
    // architecturally a NOP when ClusterID==0), backed by a device-scope
    // release/acquire atomic barrier that works under a plain launch.
    __builtin_amdgcn_s_cluster_barrier();
    if (threadIdx.x == 0) {
      __hip_atomic_fetch_add(barrier_ctr, 1u, __ATOMIC_RELEASE,
                             __HIP_MEMORY_SCOPE_AGENT);
      const unsigned target = (unsigned)NWG * (unsigned)(t + 1);
      while (__hip_atomic_load(barrier_ctr, __ATOMIC_ACQUIRE,
                               __HIP_MEMORY_SCOPE_AGENT) < target)
        __builtin_amdgcn_s_sleep(1);
    }
    __syncthreads();
  }
}

// ---------------------------------------------------------------------------
extern "C" void kernel_launch(void* const* d_in, const int* in_sizes, int n_in,
                              void* d_out, int out_size, void* d_ws, size_t ws_size,
                              hipStream_t stream) {
  (void)in_sizes; (void)n_in; (void)out_size; (void)ws_size;
  const float* u      = (const float*)d_in[0];   // [B,T,I]
  const float* w_in   = (const float*)d_in[1];   // [N,I]
  const float* w      = (const float*)d_in[2];   // [N,N]
  const float* w_bias = (const float*)d_in[3];   // [N]
  float* out = (float*)d_out;                    // [B,T,N]

  // Workspace layout:
  //   [0, 256)                    barrier counter (zeroed)
  //   [256, 256+131072)           h double buffer, 2 x [32][1024] bf16 (zeroed)
  //   [256+131072, +262144)       W_in bf16 [1024][128]
  unsigned char* ws = (unsigned char*)d_ws;
  unsigned*       barrier_ctr = (unsigned*)ws;
  unsigned short* hbuf        = (unsigned short*)(ws + 256);
  unsigned short* w_in_bf     = (unsigned short*)(ws + 256 + 2 * B_ * N_ * sizeof(unsigned short));

  (void)hipMemsetAsync(ws, 0, 256 + 2 * B_ * N_ * sizeof(unsigned short), stream);
  winConvertKernel<<<64, 256, 0, stream>>>(w_in, (unsigned*)w_in_bf);
  inputProjKernel<<<(B_ * T_ / 16) / 8, 256, 0, stream>>>(u, w_in_bf, w_bias, out);
  recurrentKernel<<<NWG, 128, 0, stream>>>(w, out, hbuf, barrier_ctr);
}
